// conv2d_block_61795989455334
// MI455X (gfx1250) — matrix operand order repaired and checked by reading, not compiled, not run on hardware
//
#include <hip/hip_runtime.h>
#include <hip/hip_bf16.h>

typedef _Float16 half_t;
typedef __attribute__((ext_vector_type(8)))  _Float16 v8h;
typedef __attribute__((ext_vector_type(16))) _Float16 v16h;
typedef __attribute__((ext_vector_type(8)))  float    v8f;

// Problem constants
#define N_BATCH 32
#define C_IN    128
#define HW      56
#define K_OUT   256
#define SPATIAL (HW*HW)            // 3136
#define N_TOTAL (N_BATCH*SPATIAL)  // 100352
#define K_RED   (C_IN*9)           // 1152
#define BN_EPS  1e-5f

// Tiling: block = 64(M) x 64(N), 8 waves, each wave 16(M) x 32(N) (2 WMMA)
#define BM 64
#define BN 64
#define BK 32
#define LDA 40   // padded LDS stride (halves) for A tile rows [m][k]
#define LDB 40   // padded LDS stride (halves) for B^T tile rows [n][k]

// Workspace layout (bytes)
#define WS_MAXABS   0        // 1 uint
#define WS_CHSUM    256      // 256 floats
#define WS_CHSSQ    1280     // 256 floats
#define WS_INV      4096     // 256 floats
#define WS_SHIFT    5120     // 256 floats
#define WS_APACK    16384    // 294912 halves = 589824 bytes
#define WS_ZERO_BYTES 16384

// ---------------------------------------------------------------------------
// 1) max |w| reduction -> ws uint (bit pattern of non-negative float)
// ---------------------------------------------------------------------------
__global__ void k_maxabs(const float* __restrict__ w, unsigned int* __restrict__ out, int n) {
  __shared__ float red[256];
  int tid = threadIdx.x;
  float m = 0.0f;
  for (int i = blockIdx.x * 256 + tid; i < n; i += gridDim.x * 256)
    m = fmaxf(m, fabsf(w[i]));
  red[tid] = m;
  __syncthreads();
  for (int s = 128; s > 0; s >>= 1) {
    if (tid < s) red[tid] = fmaxf(red[tid], red[tid + s]);
    __syncthreads();
  }
  if (tid == 0) atomicMax(out, __float_as_uint(red[0]));
}

// ---------------------------------------------------------------------------
// 2) quantize weights (4-bit symmetric, per-tensor) -> f16 A_pack[256][1152]
//    OIHW flat index == oc*1152 + (c*9 + r*3 + s) : already GEMM [M][K] layout
// ---------------------------------------------------------------------------
__global__ void k_quant(const float* __restrict__ w, const unsigned int* __restrict__ maxabs,
                        half_t* __restrict__ apack, int n) {
  int i = blockIdx.x * 256 + threadIdx.x;
  if (i >= n) return;
  float scale = __uint_as_float(*maxabs) * (1.0f / 7.0f);
  float v = w[i];
  float wq = 0.0f;
  if (scale > 0.0f) {
    float q = rintf(v / scale);
    q = fminf(fmaxf(q, -7.0f), 7.0f);
    wq = q * scale;
  }
  apack[i] = (half_t)wq;
}

// ---------------------------------------------------------------------------
// 3) implicit-GEMM conv via WMMA + fused BN-statistics epilogue
//    grid = (N_TOTAL/BN, K_OUT/BM), block = 256 (8 waves, 4Mx2N wave grid)
// ---------------------------------------------------------------------------
__global__ void __launch_bounds__(256)
k_conv_wmma(const float* __restrict__ x, const half_t* __restrict__ apack,
            float* __restrict__ out, float* __restrict__ ch_sum, float* __restrict__ ch_ssq) {
  __shared__ __align__(16) half_t As[BM * LDA];   // [m][k]
  __shared__ __align__(16) half_t Bst[BN * LDB];  // [n][k] (transposed tile)
  __shared__ float s_sum[BM];
  __shared__ float s_ssq[BM];

  const int tid  = threadIdx.x;
  const int lane = tid & 31;
  const int wv   = tid >> 5;
  const int m0   = blockIdx.y * BM;
  const int n0   = blockIdx.x * BN;
  const int m_off = (wv & 3) * 16;   // wave M offset in block tile
  const int n_off = (wv >> 2) * 32;  // wave N offset in block tile
  const int hi   = lane >> 4;        // half-wave
  const int l16  = lane & 15;

  if (tid < BM) { s_sum[tid] = 0.0f; s_ssq[tid] = 0.0f; }

  // A-tile async load mapping: one b128 (8 halves) per thread (256*8 == 64*32)
  const int ae     = tid * 8;
  const int a_mrel = ae >> 5;
  const int a_krel = ae & 31;
  const unsigned a_ldsoff = (unsigned)(uintptr_t)&As[a_mrel * LDA + a_krel];

  // Fragment source pointers in LDS (constant across K loop)
  const half_t* pa  = &As[(m_off + l16) * LDA];                 // A row, lane's M
  const half_t* pb0 = &Bst[(n_off + l16) * LDB + hi * 16];      // B^T row, N subtile 0
  const half_t* pb1 = pb0 + 16 * LDB;                           // B^T row, N subtile 1

  v8f acc0 = {};
  v8f acc1 = {};

  for (int kt = 0; kt < K_RED / BK; ++kt) {
    const int k0 = kt * BK;
    __syncthreads();  // previous tile fully consumed

    // ---- A tile: async DMA global -> LDS (ASYNCcnt path) ----
    {
      const half_t* gsrc = apack + (size_t)(m0 + a_mrel) * K_RED + k0 + a_krel;
      asm volatile("global_load_async_to_lds_b128 %0, %1, off"
                   :: "v"(a_ldsoff), "v"(gsrc)
                   : "memory");
    }

    // ---- B tile: on-the-fly im2col gather (64n x 32k), store transposed ----
    #pragma unroll
    for (int i = 0; i < 8; ++i) {
      int e     = tid * 8 + i;
      int n_rel = e & 63;
      int k_rel = e >> 6;
      int kk = k0 + k_rel;
      int c  = kk / 9;
      int rs = kk - 9 * c;
      int r  = rs / 3;
      int s  = rs - 3 * r;
      int n_g = n0 + n_rel;
      int b   = n_g / SPATIAL;
      int sp  = n_g - b * SPATIAL;
      int oh  = sp / HW;
      int ow  = sp - oh * HW;
      int ih  = oh + r - 1;
      int iw  = ow + s - 1;
      float v = 0.0f;
      if ((unsigned)ih < (unsigned)HW && (unsigned)iw < (unsigned)HW) {
        const float* px = &x[(((size_t)b * C_IN + c) * HW + ih) * HW + iw];
        v = *px;
        // prime L2 toward the next K-tile's channels (speculative hint)
        if (i == 0 && c + 4 < C_IN) __builtin_prefetch(px + 4 * SPATIAL, 0, 1);
      }
      Bst[n_rel * LDB + k_rel] = (half_t)v;
    }

    asm volatile("s_wait_asynccnt 0x0" ::: "memory");  // A tile landed in LDS
    __syncthreads();                                    // publish A+B tiles

    // ---- assemble fragments in contiguous k order (same for A and B) ----
    // A 16x32 f16: halves e0..15 -> K = hi*16 + e
    v8h alo = *(const v8h*)(pa + hi * 16);
    v8h ahi = *(const v8h*)(pa + hi * 16 + 8);
    v16h afrag = __builtin_shufflevector(alo, ahi, 0,1,2,3,4,5,6,7,8,9,10,11,12,13,14,15);
    // B 32x16 f16: halves e0..15 -> K = hi*16 + e (column-contiguous in Bst rows)
    v8h b0lo = *(const v8h*)(pb0);
    v8h b0hi = *(const v8h*)(pb0 + 8);
    v16h bfrag0 = __builtin_shufflevector(b0lo, b0hi, 0,1,2,3,4,5,6,7,8,9,10,11,12,13,14,15);
    v8h b1lo = *(const v8h*)(pb1);
    v8h b1hi = *(const v8h*)(pb1 + 8);
    v16h bfrag1 = __builtin_shufflevector(b1lo, b1hi, 0,1,2,3,4,5,6,7,8,9,10,11,12,13,14,15);

    acc0 = __builtin_amdgcn_wmma_f32_16x16x32_f16(false, afrag, false, bfrag0,
                                                  (short)0, acc0, false, false);
    acc1 = __builtin_amdgcn_wmma_f32_16x16x32_f16(false, afrag, false, bfrag1,
                                                  (short)0, acc1, false, false);
  }

  // ---- epilogue: store conv output + accumulate BN statistics ----
  // C/D layout: VGPR r -> M = r + 8*hi ; lane%16 -> N
  v8f accs[2] = { acc0, acc1 };
  #pragma unroll
  for (int j = 0; j < 2; ++j) {
    const int n_g = n0 + n_off + j * 16 + l16;
    const int b   = n_g / SPATIAL;
    const int sp  = n_g - b * SPATIAL;
    float* outp = out + (size_t)b * K_OUT * SPATIAL + sp;
    #pragma unroll
    for (int r = 0; r < 8; ++r) {
      int m_rel = m_off + r + 8 * hi;
      float v = accs[j][r];
      outp[(size_t)(m0 + m_rel) * SPATIAL] = v;
      atomicAdd(&s_sum[m_rel], v);
      atomicAdd(&s_ssq[m_rel], v * v);
    }
  }

  __syncthreads();
  if (tid < BM) {
    atomicAdd(&ch_sum[m0 + tid], s_sum[tid]);
    atomicAdd(&ch_ssq[m0 + tid], s_ssq[tid]);
  }
}

// ---------------------------------------------------------------------------
// 4) per-channel BN finalize: inv = gamma*rsqrt(var+eps), shift = beta - mean*inv
// ---------------------------------------------------------------------------
__global__ void k_bnstat(const float* __restrict__ ch_sum, const float* __restrict__ ch_ssq,
                         const float* __restrict__ gamma, const float* __restrict__ beta,
                         float* __restrict__ inv, float* __restrict__ shift) {
  int c = threadIdx.x;
  const float cnt = (float)N_TOTAL;
  float mean = ch_sum[c] / cnt;
  float var  = ch_ssq[c] / cnt - mean * mean;
  float iv   = gamma[c] * rsqrtf(var + BN_EPS);
  inv[c]   = iv;
  shift[c] = beta[c] - mean * iv;
}

// ---------------------------------------------------------------------------
// 5) BN apply + ReLU, vectorized float4 in-place on d_out
// ---------------------------------------------------------------------------
__global__ void k_bnapply(float* __restrict__ out, const float* __restrict__ inv,
                          const float* __restrict__ shift) {
  size_t i4 = ((size_t)blockIdx.x * 256 + threadIdx.x);
  size_t idx = i4 * 4;                       // SPATIAL % 4 == 0 -> same channel
  int ch = (int)((idx / SPATIAL) % K_OUT);
  float iv = inv[ch], sh = shift[ch];
  float4* p = (float4*)(out + idx);
  float4 v = *p;
  v.x = fmaxf(v.x * iv + sh, 0.0f);
  v.y = fmaxf(v.y * iv + sh, 0.0f);
  v.z = fmaxf(v.z * iv + sh, 0.0f);
  v.w = fmaxf(v.w * iv + sh, 0.0f);
  *p = v;
}

// ---------------------------------------------------------------------------
extern "C" void kernel_launch(void* const* d_in, const int* in_sizes, int n_in,
                              void* d_out, int out_size, void* d_ws, size_t ws_size,
                              hipStream_t stream) {
  const float* x     = (const float*)d_in[0];
  const float* w     = (const float*)d_in[1];
  const float* gamma = (const float*)d_in[2];
  const float* beta  = (const float*)d_in[3];
  float* out = (float*)d_out;

  char* ws = (char*)d_ws;
  unsigned int* maxabs = (unsigned int*)(ws + WS_MAXABS);
  float* ch_sum = (float*)(ws + WS_CHSUM);
  float* ch_ssq = (float*)(ws + WS_CHSSQ);
  float* inv    = (float*)(ws + WS_INV);
  float* shift  = (float*)(ws + WS_SHIFT);
  half_t* apack = (half_t*)(ws + WS_APACK);

  const int nw = K_OUT * C_IN * 9; // 294912

  // zero maxabs + channel accumulators each launch (graph-capture safe)
  hipMemsetAsync(d_ws, 0, WS_ZERO_BYTES, stream);

  k_maxabs<<<128, 256, 0, stream>>>(w, maxabs, nw);
  k_quant<<<(nw + 255) / 256, 256, 0, stream>>>(w, maxabs, apack, nw);

  dim3 grid(N_TOTAL / BN, K_OUT / BM); // (1568, 4)
  k_conv_wmma<<<grid, 256, 0, stream>>>(x, apack, out, ch_sum, ch_ssq);

  k_bnstat<<<1, K_OUT, 0, stream>>>(ch_sum, ch_ssq, gamma, beta, inv, shift);

  size_t total4 = (size_t)N_BATCH * K_OUT * SPATIAL / 4; // 6422528
  k_bnapply<<<(unsigned)(total4 / 256), 256, 0, stream>>>(out, inv, shift);
}
